// TimeAwareAttention_23802708754625
// MI455X (gfx1250) — matrix operand order repaired and checked by reading, not compiled, not run on hardware
//
#include <hip/hip_runtime.h>

// ---------------------------------------------------------------------------
// TimeAwareAttention for MI455X (gfx1250): bf16 WMMA, f32 accumulate.
//   x:(2,2048,1024) -> Q/K/V proj -> per-head attn(+time bias, mask) -> out proj
// All GEMMs via v_wmma_f32_16x16x32_bf16 with pre-swizzled operand fragments.
// GEMMs: 32x64 output strip per wave, double-buffered fragment loads.
// Attention: flash-style, ping/pong K-fragment pipeline, V loads hoisted.
// ---------------------------------------------------------------------------

typedef __attribute__((ext_vector_type(16))) __bf16 v16bf;
typedef __attribute__((ext_vector_type(8)))  float  v8f;

#define B_  2
#define S_  2048
#define D_  1024
#define H_  16
#define DK_ 64
#define MT_ (B_ * S_ / 16)     // 256 row tiles of the (4096 x 1024) matrices
#define KT_ (D_ / 32)          // 32 k tiles

__device__ __forceinline__ v8f wmma_bf16(v16bf a, v16bf b, v8f c) {
  // (neg_a, A, neg_b, B, c_mod, C, reuse_a, reuse_b)
  return __builtin_amdgcn_wmma_f32_16x16x32_bf16(false, a, false, b, (short)0, c,
                                                 false, false);
}

__device__ __forceinline__ v8f vzero8() {
  v8f r;
#pragma unroll
  for (int i = 0; i < 8; ++i) r[i] = 0.0f;
  return r;
}

// ---------------------------------------------------------------------------
// Pack x (f32 row-major 4096x1024) into bf16 A-fragments.
// A 16x32 fragment layout (ISA 7.12.2): lane half=lane/16, m=lane%16,
//   slot pair v (0..7): k = (v<4?0:16) + half*8 + (v&3)*2 (+j)
// One thread = one lane of one tile; writes 16 contiguous bf16 (32B).
// ---------------------------------------------------------------------------
__global__ void k_pack_x(const float* __restrict__ x, __bf16* __restrict__ Xa) {
  int t    = blockIdx.x * blockDim.x + threadIdx.x;  // (mt*32+kt)*32 + lane
  int lane = t & 31;
  int tile = t >> 5;
  int kt   = tile & 31;
  int mt   = tile >> 5;
  int half = lane >> 4;
  int row  = mt * 16 + (lane & 15);
  v16bf o;
#pragma unroll
  for (int v = 0; v < 8; ++v) {
    int k0 = kt * 32 + (v < 4 ? 0 : 16) + half * 8 + ((v & 3) << 1);
    float2 xv = *(const float2*)(x + (size_t)row * D_ + k0);
    o[2 * v]     = (__bf16)xv.x;
    o[2 * v + 1] = (__bf16)xv.y;
  }
  *(v16bf*)(Xa + (size_t)t * 16) = o;
}

// ---------------------------------------------------------------------------
// Pack weights (f32 1024x1024, row = input dim) into bf16 B-fragments.
// B 32x16 fragment: lane half=lane/16 -> K offset 0/8, n=lane%16,
//   VGPR v: k = (v<4?0:16) + half*8 + (v&3)*2 (+j).   blockIdx.y selects Wq/Wk/Wv/Wo.
// ---------------------------------------------------------------------------
__global__ void k_pack_w(const float* __restrict__ wq, const float* __restrict__ wk,
                         const float* __restrict__ wv, const float* __restrict__ wo,
                         __bf16* __restrict__ WbAll) {
  int z = blockIdx.y;
  const float* W = (z == 0) ? wq : (z == 1) ? wk : (z == 2) ? wv : wo;
  __bf16* dst = WbAll + (size_t)z * D_ * D_;
  int t    = blockIdx.x * blockDim.x + threadIdx.x;  // (kt*64+nt)*32 + lane
  int lane = t & 31;
  int tile = t >> 5;
  int nt   = tile & 63;
  int kt   = tile >> 6;
  int half = lane >> 4;
  int col  = nt * 16 + (lane & 15);
  v16bf o;
#pragma unroll
  for (int v = 0; v < 8; ++v) {
    int k = kt * 32 + (v < 4 ? 0 : 16) + half * 8 + ((v & 3) << 1);
    o[2 * v]     = (__bf16)W[(size_t)k * D_ + col];
    o[2 * v + 1] = (__bf16)W[(size_t)(k + 1) * D_ + col];
  }
  *(v16bf*)(dst + (size_t)t * 16) = o;
}

// ---------------------------------------------------------------------------
// Per-key additive bias: biasK[b][h][s] = time_emb[clip(td[b][s])][h], masked.
// ---------------------------------------------------------------------------
__global__ void k_bias(const int* __restrict__ td, const int* __restrict__ mask,
                       const float* __restrict__ temb, float* __restrict__ biasK) {
  int t  = blockIdx.x * blockDim.x + threadIdx.x;  // ((b*16+h)*2048)+s
  int s  = t & (S_ - 1);
  int bh = t >> 11;
  int h  = bh & 15;
  int b  = bh >> 4;
  int bin = td[b * S_ + s];
  bin = bin < 0 ? 0 : (bin > 999 ? 999 : bin);
  float v = temb[bin * H_ + h];
  if (mask[b * S_ + s] == 0) v = -1e9f;
  biasK[t] = v;
}

// ---------------------------------------------------------------------------
// Fragment-layout scatter index helpers (bf16 element index in dest buffer).
// ---------------------------------------------------------------------------
__device__ __forceinline__ size_t idx_q_afrag(int bh, int s, int dki) {
  int mt2 = s >> 4, kt2 = dki >> 5, kk = dki & 31;
  int Lp = ((kk & 15) >> 3) * 16 + (s & 15);
  int vp = (kk < 16 ? 0 : 4) + ((kk & 7) >> 1);
  return ((((size_t)bh * 128 + mt2) * 2 + kt2) * 32 + Lp) * 16 + vp * 2 + (kk & 1);
}
__device__ __forceinline__ size_t idx_k_bfrag(int bh, int s, int dki) {
  int kt2 = dki >> 5, nt2 = s >> 4, kk = dki & 31;
  int Lp = ((kk & 15) >> 3) * 16 + (s & 15);
  int vp = (kk < 16 ? 0 : 4) + ((kk & 7) >> 1);
  return ((((size_t)bh * 2 + kt2) * 128 + nt2) * 32 + Lp) * 16 + vp * 2 + (kk & 1);
}
__device__ __forceinline__ size_t idx_v_bfrag(int bh, int s, int dki) {
  int kt2 = s >> 5, nt2 = dki >> 4, kk = s & 31;
  int Lp = ((kk & 15) >> 3) * 16 + (dki & 15);
  int vp = (kk < 16 ? 0 : 4) + ((kk & 7) >> 1);
  return ((((size_t)bh * 64 + kt2) * 4 + nt2) * 32 + Lp) * 16 + vp * 2 + (kk & 1);
}

// ---------------------------------------------------------------------------
// Projection GEMM: C(4096x1024) = Xa * Wb + bias.
// Each wave -> 32x64 output strip (2 M-tiles x 4 N-tiles), double-buffered
// fragment loads, fully unrolled K loop (constant-indexed ping/pong regs).
// Scatter-store bf16 result directly into attention-ready fragment layouts:
//   mode 0: Q  -> A-fragments   Qa[b][h][mt(128)][kt(2)][lane][16]
//   mode 1: K  -> B-frag of K^T Kb[b][h][kt(2)][nt(128)][lane][16]
//   mode 2: V  -> B-fragments   Vb[b][h][kt(64)][nt(4)][lane][16]
// ---------------------------------------------------------------------------
__global__ void k_gemm_proj(const __bf16* __restrict__ Xa, const __bf16* __restrict__ Wb,
                            const float* __restrict__ bias, __bf16* __restrict__ dst,
                            int mode) {
  int lane  = threadIdx.x & 31;
  int wg    = blockIdx.x * (blockDim.x >> 5) + (threadIdx.x >> 5);  // 0..2047
  int mp    = wg >> 4;   // 0..127 -> rows mp*32 .. mp*32+31
  int ng    = wg & 15;   // 64 output cols
  int ncol  = lane & 15;
  int nhalf = lane >> 4;

  const __bf16* Abase = Xa + (((size_t)mp * 2 * KT_) * 32 + lane) * 16;
  const __bf16* Bbase = Wb + (((size_t)ng * 4) * 32 + lane) * 16;

  v8f acc[2][4];
#pragma unroll
  for (int j = 0; j < 4; ++j) {
    float bv = bias[ng * 64 + j * 16 + ncol];
#pragma unroll
    for (int v = 0; v < 8; ++v) { acc[0][j][v] = bv; acc[1][j][v] = bv; }
  }

  v16bf ab[2][2], bb[2][4];
  ab[0][0] = *(const v16bf*)(Abase);
  ab[0][1] = *(const v16bf*)(Abase + (size_t)KT_ * 512);
#pragma unroll
  for (int j = 0; j < 4; ++j) bb[0][j] = *(const v16bf*)(Bbase + (size_t)j * 512);

#pragma unroll
  for (int kt = 0; kt < KT_; ++kt) {
    const int cur = kt & 1, nxt = cur ^ 1;
    if (kt + 1 < KT_) {
      ab[nxt][0] = *(const v16bf*)(Abase + (size_t)(kt + 1) * 512);
      ab[nxt][1] = *(const v16bf*)(Abase + (size_t)(KT_ + kt + 1) * 512);
#pragma unroll
      for (int j = 0; j < 4; ++j)
        bb[nxt][j] = *(const v16bf*)(Bbase + (size_t)((kt + 1) * 64 + j) * 512);
    }
#pragma unroll
    for (int i = 0; i < 2; ++i)
#pragma unroll
      for (int j = 0; j < 4; ++j)
        acc[i][j] = wmma_bf16(ab[cur][i], bb[cur][j], acc[i][j]);
  }

#pragma unroll
  for (int i = 0; i < 2; ++i) {
    int mt = mp * 2 + i;
#pragma unroll
    for (int j = 0; j < 4; ++j) {
#pragma unroll
      for (int v = 0; v < 8; ++v) {
        float val = acc[i][j][v];
        int m_g = mt * 16 + nhalf * 8 + v;   // token row (C layout)
        int n_g = ng * 64 + j * 16 + ncol;   // feature col
        int bb2 = m_g >> 11;
        int s   = m_g & (S_ - 1);
        int hh  = n_g >> 6;
        int dki = n_g & 63;
        int bh  = bb2 * H_ + hh;
        size_t idx;
        if (mode == 0)      idx = idx_q_afrag(bh, s, dki);
        else if (mode == 1) idx = idx_k_bfrag(bh, s, dki);
        else                idx = idx_v_bfrag(bh, s, dki);
        dst[idx] = (__bf16)val;
      }
    }
  }
}

// ---------------------------------------------------------------------------
// One flash-attention step: 32 keys (group g) for one 16-query tile.
// kf[4] = K fragments for this group (preloaded by caller, ping/pong).
// ---------------------------------------------------------------------------
__device__ __forceinline__ void attn_step(
    int g, const v16bf kf[4], const v16bf qf[2], const __bf16* __restrict__ Vbh,
    const float* __restrict__ bcol, __bf16* pbuf, int lane,
    float mrow[8], float lrow[8], v8f acc[4]) {
  int nhalf = lane >> 4;
  int ncol  = lane & 15;

  // V fragments for this group: issue early, consumed after softmax
  v16bf vf[4];
#pragma unroll
  for (int j = 0; j < 4; ++j)
    vf[j] = *(const v16bf*)(Vbh + ((size_t)g * 4 + j) * 512 + lane * 16);

  // scores: two 16-key column tiles
  v8f s[2];
#pragma unroll
  for (int tc = 0; tc < 2; ++tc) {
    v8f sa = vzero8();
    sa = wmma_bf16(qf[0], kf[tc * 2 + 0], sa);
    sa = wmma_bf16(qf[1], kf[tc * 2 + 1], sa);
    float tb = bcol[g * 32 + tc * 16 + ncol];
#pragma unroll
    for (int v = 0; v < 8; ++v) sa[v] = sa[v] * 0.125f + tb;  // 1/sqrt(64)
    s[tc] = sa;
  }

  // group row-max across both tiles and the 16 lanes of this half
  float newm[8];
#pragma unroll
  for (int v = 0; v < 8; ++v) {
    float mv = fmaxf(s[0][v], s[1][v]);
    mv = fmaxf(mv, __shfl_xor(mv, 1));
    mv = fmaxf(mv, __shfl_xor(mv, 2));
    mv = fmaxf(mv, __shfl_xor(mv, 4));
    mv = fmaxf(mv, __shfl_xor(mv, 8));
    newm[v] = fmaxf(mrow[v], mv);
  }

  // p = exp(s - m); stage into per-wave LDS A-fragment (16x32).
  // Same-wave DS ops are in-order; compiler orders aliasing accesses and
  // inserts s_wait_dscnt before the consuming WMMA — no explicit fence.
#pragma unroll
  for (int tc = 0; tc < 2; ++tc) {
#pragma unroll
    for (int v = 0; v < 8; ++v) {
      float p = __expf(s[tc][v] - newm[v]);
      s[tc][v] = p;
      int mloc = v + 8 * nhalf;
      int Lp = (ncol >= 8 ? 16 : 0) + mloc;
      int vp = (tc ? 4 : 0) + ((ncol & 7) >> 1);
      pbuf[Lp * 16 + vp * 2 + (ncol & 1)] = (__bf16)p;
    }
  }

  // online-softmax state update + rescale ctx accumulators
#pragma unroll
  for (int v = 0; v < 8; ++v) {
    float rs = s[0][v] + s[1][v];
    rs += __shfl_xor(rs, 1);
    rs += __shfl_xor(rs, 2);
    rs += __shfl_xor(rs, 4);
    rs += __shfl_xor(rs, 8);
    float c = __expf(mrow[v] - newm[v]);
    lrow[v] = lrow[v] * c + rs;
    mrow[v] = newm[v];
#pragma unroll
    for (int j = 0; j < 4; ++j) acc[j][v] *= c;
  }

  v16bf pf = *(const v16bf*)(pbuf + lane * 16);
#pragma unroll
  for (int j = 0; j < 4; ++j) acc[j] = wmma_bf16(pf, vf[j], acc[j]);
}

__device__ __forceinline__ void load_k4(v16bf kf[4], const __bf16* __restrict__ Kbh,
                                        int g, int lane) {
#pragma unroll
  for (int tc = 0; tc < 2; ++tc) {
    int ntile = g * 2 + tc;
    kf[tc * 2 + 0] = *(const v16bf*)(Kbh + ((size_t)(0 * 128 + ntile)) * 512 + lane * 16);
    kf[tc * 2 + 1] = *(const v16bf*)(Kbh + ((size_t)(1 * 128 + ntile)) * 512 + lane * 16);
  }
}

// ---------------------------------------------------------------------------
// Flash attention: 1 wave = 16 queries of one (b,h); 64 key-groups of 32.
// K fragments ping/pong-pipelined across steps; ctx written directly as
// A-fragments of the (4096x1024) matrix for the output GEMM.
// ---------------------------------------------------------------------------
__global__ void k_attn(const __bf16* __restrict__ Qa, const __bf16* __restrict__ Kb,
                       const __bf16* __restrict__ Vb, const float* __restrict__ biasK,
                       __bf16* __restrict__ CtxA) {
  __shared__ __attribute__((aligned(32))) __bf16 lds[8 * 512];
  int wave = threadIdx.x >> 5;
  int lane = threadIdx.x & 31;
  int bh   = blockIdx.y;        // b*16+h
  int b    = bh >> 4;
  int h    = bh & 15;
  int qt   = blockIdx.x * 8 + wave;   // query tile 0..127
  __bf16* pbuf = &lds[wave * 512];

  const __bf16* Kbh  = Kb + (size_t)bh * 2 * 128 * 512;
  const __bf16* Vbh  = Vb + (size_t)bh * 64 * 4 * 512;
  const float*  bcol = biasK + (size_t)bh * S_;

  v16bf qf[2];
  {
    const __bf16* qb = Qa + (((size_t)(bh * 128 + qt) * 2) * 32 + lane) * 16;
    qf[0] = *(const v16bf*)(qb);
    qf[1] = *(const v16bf*)(qb + 32 * 16);
  }

  float mrow[8], lrow[8];
  v8f acc[4];
#pragma unroll
  for (int v = 0; v < 8; ++v) { mrow[v] = -3.0e38f; lrow[v] = 0.0f; }
#pragma unroll
  for (int j = 0; j < 4; ++j) acc[j] = vzero8();

  v16bf kbufA[4], kbufB[4];
  load_k4(kbufA, Kbh, 0, lane);

  for (int g = 0; g < 64; g += 2) {
    load_k4(kbufB, Kbh, g + 1, lane);
    attn_step(g, kbufA, qf, Vbh, bcol, pbuf, lane, mrow, lrow, acc);
    if (g + 2 < 64) load_k4(kbufA, Kbh, g + 2, lane);
    attn_step(g + 1, kbufB, qf, Vbh, bcol, pbuf, lane, mrow, lrow, acc);
  }

  // normalize and write ctx directly as A-fragments of the (4096x1024) matrix
  int nhalf = lane >> 4;
  int ncol  = lane & 15;
#pragma unroll
  for (int j = 0; j < 4; ++j) {
#pragma unroll
    for (int v = 0; v < 8; ++v) {
      float o = acc[j][v] / lrow[v];
      int mloc = v + 8 * nhalf;               // row within query tile
      int n_g  = h * 64 + j * 16 + ncol;      // feature col
      int mtA  = b * 128 + qt;
      int ktA  = n_g >> 5;
      int kk   = n_g & 31;
      int halfA = (kk & 15) >> 3;
      int vA = (kk < 16 ? 0 : 4) + ((kk & 7) >> 1);
      size_t idx = (((size_t)(mtA * KT_ + ktA) * 32) + halfA * 16 + mloc) * 16 +
                   vA * 2 + (kk & 1);
      CtxA[idx] = (__bf16)o;
    }
  }
}

// ---------------------------------------------------------------------------
// Output GEMM: out(4096x1024 f32) = CtxA * Wo + bo.  Same 32x64 strip/wave,
// double-buffered loads, f32 row-major stores.
// ---------------------------------------------------------------------------
__global__ void k_gemm_out(const __bf16* __restrict__ Ca, const __bf16* __restrict__ Wb,
                           const float* __restrict__ bias, float* __restrict__ out) {
  int lane  = threadIdx.x & 31;
  int wg    = blockIdx.x * (blockDim.x >> 5) + (threadIdx.x >> 5);
  int mp    = wg >> 4;
  int ng    = wg & 15;
  int ncol  = lane & 15;
  int nhalf = lane >> 4;

  const __bf16* Abase = Ca + (((size_t)mp * 2 * KT_) * 32 + lane) * 16;
  const __bf16* Bbase = Wb + (((size_t)ng * 4) * 32 + lane) * 16;

  v8f acc[2][4];
#pragma unroll
  for (int j = 0; j < 4; ++j) {
    float bv = bias[ng * 64 + j * 16 + ncol];
#pragma unroll
    for (int v = 0; v < 8; ++v) { acc[0][j][v] = bv; acc[1][j][v] = bv; }
  }

  v16bf ab[2][2], bb[2][4];
  ab[0][0] = *(const v16bf*)(Abase);
  ab[0][1] = *(const v16bf*)(Abase + (size_t)KT_ * 512);
#pragma unroll
  for (int j = 0; j < 4; ++j) bb[0][j] = *(const v16bf*)(Bbase + (size_t)j * 512);

#pragma unroll
  for (int kt = 0; kt < KT_; ++kt) {
    const int cur = kt & 1, nxt = cur ^ 1;
    if (kt + 1 < KT_) {
      ab[nxt][0] = *(const v16bf*)(Abase + (size_t)(kt + 1) * 512);
      ab[nxt][1] = *(const v16bf*)(Abase + (size_t)(KT_ + kt + 1) * 512);
#pragma unroll
      for (int j = 0; j < 4; ++j)
        bb[nxt][j] = *(const v16bf*)(Bbase + (size_t)((kt + 1) * 64 + j) * 512);
    }
#pragma unroll
    for (int i = 0; i < 2; ++i)
#pragma unroll
      for (int j = 0; j < 4; ++j)
        acc[i][j] = wmma_bf16(ab[cur][i], bb[cur][j], acc[i][j]);
  }

#pragma unroll
  for (int i = 0; i < 2; ++i) {
    int mt = mp * 2 + i;
#pragma unroll
    for (int j = 0; j < 4; ++j) {
#pragma unroll
      for (int v = 0; v < 8; ++v) {
        int m_g = mt * 16 + nhalf * 8 + v;
        int n_g = ng * 64 + j * 16 + ncol;
        out[(size_t)m_g * D_ + n_g] = acc[i][j][v];
      }
    }
  }
}

// ---------------------------------------------------------------------------
extern "C" void kernel_launch(void* const* d_in, const int* in_sizes, int n_in,
                              void* d_out, int out_size, void* d_ws, size_t ws_size,
                              hipStream_t stream) {
  const float* x    = (const float*)d_in[0];
  const int*   td   = (const int*)d_in[1];
  const int*   mask = (const int*)d_in[2];
  const float* Wq   = (const float*)d_in[3];
  const float* bq   = (const float*)d_in[4];
  const float* Wk   = (const float*)d_in[5];
  const float* bk   = (const float*)d_in[6];
  const float* Wv   = (const float*)d_in[7];
  const float* bv   = (const float*)d_in[8];
  const float* temb = (const float*)d_in[9];
  const float* Wo   = (const float*)d_in[10];
  const float* bo   = (const float*)d_in[11];
  float* out = (float*)d_out;

  const size_t NMAT = (size_t)B_ * S_ * D_;   // 4096*1024 elements
  __bf16* xa    = (__bf16*)d_ws;              // 8 MB
  __bf16* wb    = xa + NMAT;                  // 4 x 2 MB (q,k,v,o)
  __bf16* qa    = wb + (size_t)4 * D_ * D_;   // 8 MB
  __bf16* kb    = qa + NMAT;                  // 8 MB
  __bf16* vbuf  = kb + NMAT;                  // 8 MB
  __bf16* ca    = vbuf + NMAT;                // 8 MB
  float*  biasK = (float*)(ca + NMAT);        // 256 KB

  k_pack_x<<<dim3(NMAT / 16 / 256), 256, 0, stream>>>(x, xa);
  k_pack_w<<<dim3((D_ * D_ / 16) / 256, 4), 256, 0, stream>>>(Wq, Wk, Wv, Wo, wb);
  k_bias<<<dim3(B_ * H_ * S_ / 256), 256, 0, stream>>>(td, mask, temb, biasK);

  // 2048 waves per GEMM -> 256 blocks of 8 waves
  k_gemm_proj<<<dim3(256), 256, 0, stream>>>(xa, wb + 0 * D_ * D_, bq, qa, 0);
  k_gemm_proj<<<dim3(256), 256, 0, stream>>>(xa, wb + 1 * D_ * D_, bk, kb, 1);
  k_gemm_proj<<<dim3(256), 256, 0, stream>>>(xa, wb + 2 * D_ * D_, bv, vbuf, 2);

  k_attn<<<dim3(16, B_ * H_), 256, 0, stream>>>(qa, kb, vbuf, biasK, ca);

  k_gemm_out<<<dim3(256), 256, 0, stream>>>(ca, wb + 3 * (size_t)D_ * D_, bo, out);
}
